// BiGRU_63969242906953
// MI455X (gfx1250) — matrix operand order repaired and checked by reading, not compiled, not run on hardware
//
#include <hip/hip_runtime.h>
#include <hip/hip_bf16.h>

// ---------------------------------------------------------------------------
// BiGRU + CRF for MI455X (gfx1250, wave32, WMMA bf16 16x16x32,
// async-to-LDS staging, TDM xp prefetch in the recurrent loop)
// B=64 T=512 V=21128 E=256 H=512 K=31, 3H=1536, M=B*T=32768
// ---------------------------------------------------------------------------

typedef __bf16 bf16_t;
typedef __bf16 bf16x8  __attribute__((ext_vector_type(8)));
typedef __bf16 bf16x16 __attribute__((ext_vector_type(16)));
typedef float  f32x8   __attribute__((ext_vector_type(8)));
typedef unsigned int u32x4 __attribute__((ext_vector_type(4)));
typedef int          i32x8 __attribute__((ext_vector_type(8)));
typedef int          i32x4 __attribute__((ext_vector_type(4)));

#define WMMA_BF16(a, b, c) \
  __builtin_amdgcn_wmma_f32_16x16x32_bf16(false, (a), false, (b), (short)0, (c), false, false)

__device__ __forceinline__ bf16x16 join16(bf16x8 lo, bf16x8 hi) {
  bf16x16 r;
#pragma unroll
  for (int i = 0; i < 8; ++i) { r[i] = lo[i]; r[i + 8] = hi[i]; }
  return r;
}

// LDS byte offset of a generic pointer to __shared__ (flat = aperture | offset)
__device__ __forceinline__ uint32_t lds_off(const void* p) {
  return (uint32_t)(uintptr_t)p;
}

// GLOBAL_LOAD_ASYNC_TO_LDS_B128: memory -> LDS, no VGPR transit, ASYNCcnt
__device__ __forceinline__ void async_ld_b128(uint32_t ldsoff, const void* g) {
  asm volatile("global_load_async_to_lds_b128 %0, %1, off"
               :: "v"(ldsoff), "v"(g) : "memory");
}
__device__ __forceinline__ void wait_async0() {
  asm volatile("s_wait_asynccnt 0" ::: "memory");
}

// TDM 2D tile load: tile_dim0=1536 f32, tile_dim1=16 rows, row stride T*3H.
// D# packed per CDNA5 ISA sec 8.3/8.4; this toolchain's builtin takes 6 args
// (g0, g1, g2, g3, extra group, cpol).
__device__ __forceinline__ void tdm_load_xp_tile(uint32_t ldsoff, const void* gaddr) {
  const uint64_t ga = (uint64_t)(uintptr_t)gaddr;
  u32x4 g0;
  g0[0] = 1u;                                   // count=1 (user descriptor)
  g0[1] = ldsoff;                               // lds_addr
  g0[2] = (uint32_t)ga;                         // global_addr[31:0]
  g0[3] = (uint32_t)((ga >> 32) & 0x01FFFFFFull) | 0x80000000u;  // addr[56:32] | type=2
  i32x8 g1;
  g1[0] = 0x20000;                              // data_size=2 (4 bytes)
  g1[1] = (int)(1536u << 16);                   // tensor_dim0 = 1536 (bits 79:48 lo)
  g1[2] = (int)0x80000000u;                     // tensor_dim1 = 32768 (bits 111:80 lo)
  g1[3] = (int)(1536u << 16);                   // tile_dim0 = 1536 (bits 127:112)
  g1[4] = 16;                                   // tile_dim1 = 16, tile_dim2 = 0
  g1[5] = 786432;                               // tensor_dim0_stride = T*3H (lo32)
  g1[6] = 0;
  g1[7] = 0;
  i32x4 g2 = {0, 0, 0, 0};
  i32x4 g3 = {0, 0, 0, 0};
  i32x8 g4 = {0, 0, 0, 0, 0, 0, 0, 0};
  __builtin_amdgcn_tensor_load_to_lds(g0, g1, g2, g3, g4, 0);
}

// --------------------------- elementwise helpers ---------------------------

__global__ void cast_pad_kernel(const float* __restrict__ src, bf16_t* __restrict__ dst,
                                int n, int ntot) {
  int i = blockIdx.x * blockDim.x + threadIdx.x;
  if (i < ntot) dst[i] = (i < n) ? (bf16_t)src[i] : (bf16_t)0.0f;
}

__global__ void embed_kernel(const int* __restrict__ ids, const float* __restrict__ emb,
                             bf16_t* __restrict__ x) {
  const int bt = blockIdx.x;
  const int e  = threadIdx.x;
  const int id = ids[bt];
  x[(size_t)bt * 256 + e] = (bf16_t)emb[(size_t)id * 256 + e];
}

__global__ void zero_loss_kernel(float* loss) { *loss = 0.0f; }

// ------------------------------- WMMA GEMM ---------------------------------
// C[M x Nreal] = A[M x K](bf16) * W[Npad x K](bf16)^T + bias.
// Block tile 128x64, 8 waves. LDS double-buffered, filled with
// global_load_async_to_lds_b128 so DMA overlaps WMMA (1 barrier / K-chunk).
__global__ __launch_bounds__(256) void gemm_bf16_kernel(
    const bf16_t* __restrict__ A, const bf16_t* __restrict__ W,
    const float* __restrict__ bias, float* __restrict__ C,
    int K, int ldc, int Nreal) {
  __shared__ bf16_t As[2][128 * 72];
  __shared__ bf16_t Bs[2][64 * 72];

  const int tid  = threadIdx.x;
  const int lane = tid & 31;
  const int wave = tid >> 5;
  const int m0   = blockIdx.x * 128;
  const int n0   = blockIdx.y * 64;

  const uint32_t asB = lds_off(&As[0][0]);
  const uint32_t bsB = lds_off(&Bs[0][0]);

  auto stage = [&](int kc, int buf) {
    const int k0 = kc << 6;
#pragma unroll
    for (int i = 0; i < 4; ++i) {   // A tile: 128x64, 4 x b128 per thread
      const int cid = i * 256 + tid;
      const int r   = cid >> 3;
      const int c8  = (cid & 7) << 3;
      async_ld_b128(asB + (uint32_t)(buf * 128 * 72 + r * 72 + c8) * 2,
                    &A[(size_t)(m0 + r) * K + k0 + c8]);
    }
#pragma unroll
    for (int i = 0; i < 2; ++i) {   // B tile: 64x64, 2 x b128 per thread
      const int cid = i * 256 + tid;
      const int r   = cid >> 3;
      const int c8  = (cid & 7) << 3;
      async_ld_b128(bsB + (uint32_t)(buf * 64 * 72 + r * 72 + c8) * 2,
                    &W[(size_t)(n0 + r) * K + k0 + c8]);
    }
  };

  f32x8 acc[4];
#pragma unroll
  for (int t = 0; t < 4; ++t)
#pragma unroll
    for (int r = 0; r < 8; ++r) acc[t][r] = 0.0f;

  const int nchunk = K >> 6;
  stage(0, 0);
  wait_async0();
  __syncthreads();

  const int arow  = (wave * 16 + (lane & 15)) * 72;
  const int khalf = (lane >> 4) << 3;   // A: +8 k for lanes 16..31
  const int koff  = (lane >> 4) << 4;   // +16 for lanes 16..31, unused (B uses khalf)

  for (int kc = 0; kc < nchunk; ++kc) {
    const int buf = kc & 1;
    if (kc + 1 < nchunk) stage(kc + 1, buf ^ 1);  // DMA next chunk during compute
    if (kc + 2 < nchunk)                          // L2 prefetch two chunks ahead
      __builtin_prefetch(&A[(size_t)(m0 + (tid >> 1)) * K + ((kc + 2) << 6)], 0, 1);

#pragma unroll
    for (int kb = 0; kb < 64; kb += 32) {
      const bf16x16 af = join16(*(const bf16x8*)&As[buf][arow + kb + khalf],
                                *(const bf16x8*)&As[buf][arow + kb + 16 + khalf]);
#pragma unroll
      for (int nt = 0; nt < 4; ++nt) {
        const int brow = (nt * 16 + (lane & 15)) * 72;
        const bf16x16 bfr = join16(*(const bf16x8*)&Bs[buf][brow + kb + khalf],
                                   *(const bf16x8*)&Bs[buf][brow + kb + 16 + khalf]);
        acc[nt] = WMMA_BF16(af, bfr, acc[nt]);
      }
    }
    wait_async0();     // own async writes to the other buffer done
    __syncthreads();   // everyone done reading buf / writing buf^1
  }

  const int mrow = m0 + wave * 16 + ((lane >> 4) << 3);
#pragma unroll
  for (int nt = 0; nt < 4; ++nt) {
    const int col = n0 + nt * 16 + (lane & 15);
    if (col < Nreal) {
      const float bv = bias ? bias[col] : 0.0f;
#pragma unroll
      for (int r = 0; r < 8; ++r)
        C[(size_t)(mrow + r) * ldc + col] = acc[nt][r] + bv;
    }
  }
}

// ---------------------------- recurrent GRU layer --------------------------
// Grid (4 batch-tiles, 2 dirs). 16 batch rows/block, h resident in LDS,
// gh = h(16x512) @ w_hh^T(512x1536) via WMMA each step, fused gates after.
// xp tiles (16x1536 f32) are DMA'd by the Tensor Data Mover into an LDS
// double buffer one step ahead (TENSORcnt-synchronized).
__global__ __launch_bounds__(256) void gru_layer_kernel(
    const float* __restrict__ xp_f, const float* __restrict__ xp_b,
    const bf16_t* __restrict__ whh_f, const bf16_t* __restrict__ whh_b,
    const float* __restrict__ bhh_f, const float* __restrict__ bhh_b,
    bf16_t* __restrict__ out) {
  extern __shared__ char smem[];
  bf16_t* hbuf  = (bf16_t*)smem;                             // [16][520] bf16
  float*  gh    = (float*)(smem + 16 * 520 * 2);             // [16][1536] f32
  float*  xpbuf = (float*)(smem + 16 * 520 * 2 + 16 * 1536 * 4);  // 2x[16][1536] f32

  const int dir = blockIdx.y;
  const int b0  = blockIdx.x * 16;
  const float*  xp  = dir ? xp_b  : xp_f;
  const bf16_t* whh = dir ? whh_b : whh_f;
  const float*  bhh = dir ? bhh_b : bhh_f;

  const int tid  = threadIdx.x;
  const int lane = tid & 31;
  const int wave = tid >> 5;
  const int HS   = 520;

  const uint32_t xpB = lds_off(xpbuf);

  for (int i = tid; i < 16 * HS; i += 256) hbuf[i] = (bf16_t)0.0f;

  // TDM prefetch of step 0's xp tile into buffer 0
  if (wave == 0) {
    const int tt0 = dir ? 511 : 0;
    tdm_load_xp_tile(xpB, xp + ((size_t)b0 * 512 + tt0) * 1536);
  }
  __syncthreads();

  const int arow  = (lane & 15) * HS;
  const int khalf = (lane >> 4) << 3;
  const int koff  = (lane >> 4) << 4;
  const int mr    = (lane >> 4) << 3;

  for (int step = 0; step < 512; ++step) {
    const int tt = dir ? (511 - step) : step;

    // DMA next step's xp tile into the other buffer while we do WMMA
    if (wave == 0 && step + 1 < 512) {
      const int ttn = dir ? (511 - (step + 1)) : (step + 1);
      tdm_load_xp_tile(xpB + (uint32_t)((step + 1) & 1) * 16 * 1536 * 4,
                       xp + ((size_t)b0 * 512 + ttn) * 1536);
    }

    // gh = h @ whh^T + bhh   (8 waves x 12 N-tiles x 16 K-steps)
    for (int ntl = 0; ntl < 12; ++ntl) {
      const int ncol = wave * 192 + ntl * 16 + (lane & 15);
      const float bb = bhh[ncol];
      f32x8 acc;
#pragma unroll
      for (int r = 0; r < 8; ++r) acc[r] = bb;
      const bf16_t* wrow = whh + (size_t)ncol * 512;
#pragma unroll 4
      for (int kb = 0; kb < 512; kb += 32) {
        const bf16x16 af  = join16(*(const bf16x8*)&hbuf[arow + kb + khalf],
                                   *(const bf16x8*)&hbuf[arow + kb + 16 + khalf]);
        const bf16x16 bfr = join16(*(const bf16x8*)&wrow[kb + khalf],
                                   *(const bf16x8*)&wrow[kb + 16 + khalf]);
        acc = WMMA_BF16(af, bfr, acc);
      }
#pragma unroll
      for (int r = 0; r < 8; ++r) gh[(mr + r) * 1536 + ncol] = acc[r];
    }

    // current step's tile must be in LDS; keep next step's DMA in flight
    if (wave == 0) {
      if (step + 1 < 512) __builtin_amdgcn_s_wait_tensorcnt((short)1);
      else                __builtin_amdgcn_s_wait_tensorcnt((short)0);
    }
    __syncthreads();

    // fused GRU gates from LDS xp tile
    const float* xpt = xpbuf + (step & 1) * 16 * 1536;
    for (int idx = tid; idx < 16 * 512; idx += 256) {
      const int row = idx >> 9;
      const int j   = idx & 511;
      const float xr = xpt[row * 1536 + j];
      const float xz = xpt[row * 1536 + 512 + j];
      const float xn = xpt[row * 1536 + 1024 + j];
      const float hr = gh[row * 1536 + j];
      const float hz = gh[row * 1536 + 512 + j];
      const float hn = gh[row * 1536 + 1024 + j];
      const float hp = (float)hbuf[row * HS + j];
      const float rg = 1.0f / (1.0f + __expf(-(xr + hr)));
      const float zg = 1.0f / (1.0f + __expf(-(xz + hz)));
      const float ng = tanhf(xn + rg * hn);
      const float hnew = (1.0f - zg) * ng + zg * hp;
      hbuf[row * HS + j] = (bf16_t)hnew;
      out[((size_t)(b0 + row) * 512 + tt) * 1024 + dir * 512 + j] = (bf16_t)hnew;
    }
    __syncthreads();
  }
}

// --------------------------------- CRF -------------------------------------
__global__ __launch_bounds__(32) void crf_kernel(
    const int* __restrict__ labels, const float* __restrict__ em,
    const float* __restrict__ start, const float* __restrict__ endv,
    const float* __restrict__ trans, float* __restrict__ loss) {
  __shared__ float tr[31 * 32];
  const int b    = blockIdx.x;
  const int lane = threadIdx.x;

  for (int i = lane; i < 31 * 31; i += 32) tr[(i / 31) * 32 + (i % 31)] = trans[i];
  __syncthreads();

  const int*   lab = labels + b * 512;
  const float* emB = em + (size_t)b * 512 * 31;

  float np  = 0.0f;
  int   cnt = 0;
  for (int t = lane; t < 512; t += 32)
    if (lab[t] > -1) cnt++;
  for (int t = 1 + lane; t < 512; t += 32) {
    if (lab[t] > -1) np += tr[lab[t - 1] * 32 + lab[t]] + emB[t * 31 + lab[t]];
  }
#pragma unroll
  for (int o = 16; o > 0; o >>= 1) {
    np  += __shfl_xor(np, o, 32);
    cnt += __shfl_xor(cnt, o, 32);
  }

  const int j  = lane;
  const int jj = (j < 31) ? j : 0;
  float alpha = (j < 31) ? (start[j] + emB[j]) : -INFINITY;
  for (int t = 1; t < 512; ++t) {
    const bool mk = lab[t] > -1;
    float mmax = -INFINITY, ssum = 0.0f;
    for (int i = 0; i < 31; ++i) {
      const float ai = __shfl(alpha, i, 32);
      const float v  = ai + tr[i * 32 + jj];
      const float nm = fmaxf(mmax, v);
      ssum = ssum * __expf(mmax - nm) + __expf(v - nm);
      mmax = nm;
    }
    const float nxt = mmax + __logf(ssum) + emB[t * 31 + jj];
    if (mk && j < 31) alpha = nxt;
  }

  float v = alpha + ((j < 31) ? endv[j] : 0.0f);
  float mx = v;
#pragma unroll
  for (int o = 16; o > 0; o >>= 1) mx = fmaxf(mx, __shfl_xor(mx, o, 32));
  float se = __expf(v - mx);
#pragma unroll
  for (int o = 16; o > 0; o >>= 1) se += __shfl_xor(se, o, 32);
  const float denom = mx + __logf(se);

  if (lane == 0) {
    const float num = start[lab[0]] + emB[lab[0]] + np + endv[lab[cnt - 1]];
    atomicAdd(loss, denom - num);
  }
}

// ------------------------------- host side ---------------------------------

extern "C" void kernel_launch(void* const* d_in, const int* in_sizes, int n_in,
                              void* d_out, int out_size, void* d_ws, size_t ws_size,
                              hipStream_t stream) {
  (void)in_sizes; (void)n_in; (void)out_size; (void)ws_size;

  const int*   ids       = (const int*)d_in[0];
  const int*   labels    = (const int*)d_in[1];
  const float* emb       = (const float*)d_in[2];
  const float* w_ih_l0f  = (const float*)d_in[3];
  const float* w_hh_l0f  = (const float*)d_in[4];
  const float* b_ih_l0f  = (const float*)d_in[5];
  const float* b_hh_l0f  = (const float*)d_in[6];
  const float* w_ih_l0b  = (const float*)d_in[7];
  const float* w_hh_l0b  = (const float*)d_in[8];
  const float* b_ih_l0b  = (const float*)d_in[9];
  const float* b_hh_l0b  = (const float*)d_in[10];
  const float* w_ih_l1f  = (const float*)d_in[11];
  const float* w_hh_l1f  = (const float*)d_in[12];
  const float* b_ih_l1f  = (const float*)d_in[13];
  const float* b_hh_l1f  = (const float*)d_in[14];
  const float* w_ih_l1b  = (const float*)d_in[15];
  const float* w_hh_l1b  = (const float*)d_in[16];
  const float* b_ih_l1b  = (const float*)d_in[17];
  const float* b_hh_l1b  = (const float*)d_in[18];
  const float* cls_w     = (const float*)d_in[19];
  const float* cls_b     = (const float*)d_in[20];
  const float* crf_start = (const float*)d_in[21];
  const float* crf_end   = (const float*)d_in[22];
  const float* crf_trans = (const float*)d_in[23];

  char* ws = (char*)d_ws;
  size_t off = 0;
  auto take = [&](size_t bytes) -> char* {
    char* p = ws + off;
    off = (off + bytes + 255) & ~(size_t)255;
    return p;
  };

  bf16_t* X     = (bf16_t*)take((size_t)64 * 512 * 256 * 2);
  bf16_t* HIN   = (bf16_t*)take((size_t)64 * 512 * 1024 * 2);
  bf16_t* GOUT  = (bf16_t*)take((size_t)64 * 512 * 1024 * 2);
  float*  XPF   = (float*) take((size_t)64 * 512 * 1536 * 4);
  float*  XPB   = (float*) take((size_t)64 * 512 * 1536 * 4);
  bf16_t* WIH0F = (bf16_t*)take((size_t)1536 * 256 * 2);
  bf16_t* WIH0B = (bf16_t*)take((size_t)1536 * 256 * 2);
  bf16_t* WHH0F = (bf16_t*)take((size_t)1536 * 512 * 2);
  bf16_t* WHH0B = (bf16_t*)take((size_t)1536 * 512 * 2);
  bf16_t* WIH1F = (bf16_t*)take((size_t)1536 * 1024 * 2);
  bf16_t* WIH1B = (bf16_t*)take((size_t)1536 * 1024 * 2);
  bf16_t* WHH1F = (bf16_t*)take((size_t)1536 * 512 * 2);
  bf16_t* WHH1B = (bf16_t*)take((size_t)1536 * 512 * 2);
  bf16_t* CLSW  = (bf16_t*)take((size_t)64 * 1024 * 2);

  auto cast = [&](const float* s, bf16_t* d, int n, int ntot) {
    cast_pad_kernel<<<(ntot + 255) / 256, 256, 0, stream>>>(s, d, n, ntot);
  };
  cast(w_ih_l0f, WIH0F, 1536 * 256, 1536 * 256);
  cast(w_ih_l0b, WIH0B, 1536 * 256, 1536 * 256);
  cast(w_hh_l0f, WHH0F, 1536 * 512, 1536 * 512);
  cast(w_hh_l0b, WHH0B, 1536 * 512, 1536 * 512);
  cast(w_ih_l1f, WIH1F, 1536 * 1024, 1536 * 1024);
  cast(w_ih_l1b, WIH1B, 1536 * 1024, 1536 * 1024);
  cast(w_hh_l1f, WHH1F, 1536 * 512, 1536 * 512);
  cast(w_hh_l1b, WHH1B, 1536 * 512, 1536 * 512);
  cast(cls_w,    CLSW,  31 * 1024,  64 * 1024);

  embed_kernel<<<64 * 512, 256, 0, stream>>>(ids, emb, X);

  const dim3 gXP(32768 / 128, 1536 / 64);
  gemm_bf16_kernel<<<gXP, 256, 0, stream>>>(X, WIH0F, b_ih_l0f, XPF, 256, 1536, 1536);
  gemm_bf16_kernel<<<gXP, 256, 0, stream>>>(X, WIH0B, b_ih_l0b, XPB, 256, 1536, 1536);

  const int GRU_LDS = 16 * 520 * 2 + 16 * 1536 * 4 + 2 * 16 * 1536 * 4;  // 311,552 B
  (void)hipFuncSetAttribute((const void*)gru_layer_kernel,
                            hipFuncAttributeMaxDynamicSharedMemorySize, GRU_LDS);

  gru_layer_kernel<<<dim3(4, 2), 256, GRU_LDS, stream>>>(
      XPF, XPB, WHH0F, WHH0B, b_hh_l0f, b_hh_l0b, HIN);

  gemm_bf16_kernel<<<gXP, 256, 0, stream>>>(HIN, WIH1F, b_ih_l1f, XPF, 1024, 1536, 1536);
  gemm_bf16_kernel<<<gXP, 256, 0, stream>>>(HIN, WIH1B, b_ih_l1b, XPB, 1024, 1536, 1536);

  gru_layer_kernel<<<dim3(4, 2), 256, GRU_LDS, stream>>>(
      XPF, XPB, WHH1F, WHH1B, b_hh_l1f, b_hh_l1b, GOUT);

  float* outf = (float*)d_out;
  gemm_bf16_kernel<<<dim3(32768 / 128, 1), 256, 0, stream>>>(
      GOUT, CLSW, cls_b, outf + 1, 1024, 31, 31);

  zero_loss_kernel<<<1, 1, 0, stream>>>(outf);
  crf_kernel<<<64, 32, 0, stream>>>(labels, outf + 1, crf_start, crf_end, crf_trans, outf);
}
